// GCN_23304492548302
// MI455X (gfx1250) — matrix operand order repaired and checked by reading, not compiled, not run on hardware
//
#include <hip/hip_runtime.h>

typedef __attribute__((ext_vector_type(16))) _Float16 v16h;
typedef __attribute__((ext_vector_type(8)))  _Float16 v8h;
typedef __attribute__((ext_vector_type(4)))  _Float16 v4h;
typedef __attribute__((ext_vector_type(8)))  float    v8f;

#define TILE_M 64
#define TILE_N 64
#define TILE_K 32
#define LDS_PITCH (TILE_K + 8)   // 40 f16 = 80 bytes: multiple of 16B -> aligned b128 runs

// ---------------------------------------------------------------------------
// Generic fp32 GEMM via f16 WMMA with fp32 accumulation.
// C[M x N] (row stride ldc) = A[M x K] (row stride lda) @ B[K x N] (+ bias)
// Block = 128 threads = 4 waves; each wave owns a 16(M) x 64(N) strip.
// Fragments are read from LDS as 16-byte ds_load_b128 runs (ISA 7.12.2 layout).
// ---------------------------------------------------------------------------
__global__ __launch_bounds__(128) void gcn_gemm_wmma(
    const float* __restrict__ A, int lda,
    const float* __restrict__ B,
    const float* __restrict__ bias,          // may be nullptr
    float* __restrict__ C, int ldc,
    int M, int N, int K)
{
    __shared__ __align__(16) _Float16 As [TILE_M][LDS_PITCH];  // row-major A tile
    __shared__ __align__(16) _Float16 BsT[TILE_N][LDS_PITCH];  // B tile transposed: [col][k]

    const int tid   = threadIdx.x;
    const int wave  = tid >> 5;
    const int lane  = tid & 31;
    const int lhalf = lane >> 4;   // 0: lanes 0-15, 1: lanes 16-31
    const int l15   = lane & 15;

    const int row0 = blockIdx.y * TILE_M;
    const int col0 = blockIdx.x * TILE_N;

    v8f acc[4];
    #pragma unroll
    for (int j = 0; j < 4; ++j)
        #pragma unroll
        for (int r = 0; r < 8; ++r) acc[j][r] = 0.0f;

    for (int k0 = 0; k0 < K; k0 += TILE_K) {
        // ---- A tile (64x32): float4 global loads, packed v4h LDS stores.
        // i = tid*4 -> c is a multiple of 4, rows/cols coalesced.
        for (int i = tid * 4; i < TILE_M * TILE_K; i += 128 * 4) {
            int r = i / TILE_K, c = i % TILE_K;
            int gr = row0 + r;
            float4 v = make_float4(0.f, 0.f, 0.f, 0.f);
            if (gr < M) {
                const float* ap = A + (long)gr * lda + k0 + c;
                v = *(const float4*)ap;
                if (k0 + TILE_K < K)
                    __builtin_prefetch(ap + TILE_K, 0, 0);  // next K-tile -> global_prefetch_b8
            }
            v4h hv;
            hv[0] = (_Float16)v.x; hv[1] = (_Float16)v.y;
            hv[2] = (_Float16)v.z; hv[3] = (_Float16)v.w;
            *(v4h*)&As[r][c] = hv;
        }
        // ---- B tile (32x64) stored transposed; guard N (head GEMM has N=40)
        for (int i = tid; i < TILE_K * TILE_N; i += 128) {
            int r = i / TILE_N, c = i % TILE_N;
            int gc = col0 + c;
            float v = (gc < N) ? B[(long)(k0 + r) * N + gc] : 0.0f;
            BsT[c][r] = (_Float16)v;
        }
        __syncthreads();

        // A fragment (16-bit A 16x32, ISA 7.12.2): lane<16 holds K {0..7,16..23},
        // lane>=16 holds K {8..15,24..31} -> two contiguous 16B runs.
        const _Float16* ap = &As[wave * 16 + l15][lhalf * 8];
        v8h alo = *(const v8h*)ap;
        v8h ahi = *(const v8h*)(ap + 16);
        v16h afrag = __builtin_shufflevector(alo, ahi,
            0, 1, 2, 3, 4, 5, 6, 7, 8, 9, 10, 11, 12, 13, 14, 15);

        #pragma unroll
        for (int j = 0; j < 4; ++j) {
            // B fragment: lane = column, lane<16 holds K {0..7,16..23},
            // lane>=16 holds K {8..15,24..31} -> two contiguous 16B runs.
            const _Float16* bp = &BsT[j * 16 + l15][lhalf * 8];
            v8h blo = *(const v8h*)bp;
            v8h bhi = *(const v8h*)(bp + 16);
            v16h bfrag = __builtin_shufflevector(blo, bhi,
                0, 1, 2, 3, 4, 5, 6, 7, 8, 9, 10, 11, 12, 13, 14, 15);

            acc[j] = __builtin_amdgcn_wmma_f32_16x16x32_f16(
                false, afrag, false, bfrag, (short)0, acc[j], false, false);
        }
        __syncthreads();
    }

    // Store: C/D 16x16 f32 layout: VGPR r -> M = r + 8*lhalf, N = lane&15
    #pragma unroll
    for (int j = 0; j < 4; ++j) {
        int col = col0 + j * 16 + l15;
        if (col >= N) continue;
        float bv = bias ? bias[col] : 0.0f;
        #pragma unroll
        for (int r = 0; r < 8; ++r) {
            int row = row0 + wave * 16 + lhalf * 8 + r;
            if (row < M) C[(long)row * ldc + col] = acc[j][r] + bv;
        }
    }
}

// ---------------------------------------------------------------------------
__global__ void gcn_fill_zero(float* __restrict__ p, long n)
{
    long i = (long)blockIdx.x * blockDim.x + threadIdx.x;
    long stride = (long)gridDim.x * blockDim.x;
    for (; i < n; i += stride) p[i] = 0.0f;
}

__global__ void gcn_degree(const int* __restrict__ dst,
                           const float* __restrict__ w,
                           int E, float* __restrict__ degw,
                           float* __restrict__ deg1)
{
    int e = blockIdx.x * blockDim.x + threadIdx.x;
    int stride = gridDim.x * blockDim.x;
    for (; e < E; e += stride) {
        int d = dst[e];
        atomicAdd(&degw[d], w[e]);
        atomicAdd(&deg1[d], 1.0f);
    }
}

__global__ void gcn_dinv(const float* __restrict__ degw,
                         const float* __restrict__ deg1,
                         float* __restrict__ dinvw,
                         float* __restrict__ dinv1, int n)
{
    int i = blockIdx.x * blockDim.x + threadIdx.x;
    if (i < n) {
        float a = degw[i] + 1.0f;   // + self loop (weight 1)
        float b = deg1[i] + 1.0f;
        dinvw[i] = (a > 0.0f) ? rsqrtf(a) : 0.0f;
        dinv1[i] = (b > 0.0f) ? rsqrtf(b) : 0.0f;
    }
}

// One wave per edge: lanes stream features as float4 (512B/wave per access),
// scatter with global float atomics. Prefetch next edge's source row.
__global__ __launch_bounds__(256) void gcn_aggregate(
    const float* __restrict__ h,
    const int* __restrict__ src, const int* __restrict__ dst,
    const float* __restrict__ w,            // nullptr -> unit weights
    const float* __restrict__ dinv,
    float* __restrict__ agg, int F, int E)
{
    int gw   = (blockIdx.x * blockDim.x + threadIdx.x) >> 5;
    int lane = threadIdx.x & 31;
    int nw   = (gridDim.x * blockDim.x) >> 5;
    const int F4 = F >> 2;
    for (int e = gw; e < E; e += nw) {
        int s = src[e], d = dst[e];
        float ww   = w ? w[e] : 1.0f;
        float norm = dinv[s] * ww * dinv[d];
        const float4* hs = (const float4*)(h + (long)s * F);
        float*        ad = agg + (long)d * F;

        int en = e + nw;                      // prefetch next gather row
        if (en < E)
            __builtin_prefetch(h + (long)src[en] * F + lane * 4, 0, 0);

        for (int f = lane; f < F4; f += 32) {
            float4 hv = hs[f];
            int b = f << 2;
            atomicAdd(&ad[b + 0], norm * hv.x);
            atomicAdd(&ad[b + 1], norm * hv.y);
            atomicAdd(&ad[b + 2], norm * hv.z);
            atomicAdd(&ad[b + 3], norm * hv.w);
        }
    }
}

// out[i, f(+off)] = act( agg[i,f] + dinv[i]^2 * h[i,f] + bias[f] )
__global__ void gcn_finalize(const float* __restrict__ agg,
                             const float* __restrict__ h,
                             const float* __restrict__ dinv,
                             const float* __restrict__ bias,
                             float* __restrict__ out, int ldo,
                             int F, int n, int relu)
{
    long idx = (long)blockIdx.x * blockDim.x + threadIdx.x;
    long total = (long)n * F;
    long stride = (long)gridDim.x * blockDim.x;
    for (; idx < total; idx += stride) {
        int i = (int)(idx / F);
        int f = (int)(idx % F);
        float di = dinv[i];
        float v = agg[idx] + di * di * h[idx] + bias[f];
        if (relu) v = fmaxf(v, 0.0f);
        out[(long)i * ldo + f] = v;
    }
}

// One wave per row; C <= 64 classes (here 40).
__global__ __launch_bounds__(256) void gcn_log_softmax(float* __restrict__ logits,
                                                       int M, int C)
{
    int row  = (blockIdx.x * blockDim.x + threadIdx.x) >> 5;
    int lane = threadIdx.x & 31;
    int nw   = (gridDim.x * blockDim.x) >> 5;
    for (; row < M; row += nw) {
        float* p = logits + (long)row * C;
        float v0 = (lane      < C) ? p[lane]      : -1e30f;
        float v1 = (lane + 32 < C) ? p[lane + 32] : -1e30f;
        float m = fmaxf(v0, v1);
        #pragma unroll
        for (int off = 16; off; off >>= 1)
            m = fmaxf(m, __shfl_xor(m, off, 32));
        float s = 0.0f;
        if (lane      < C) s += expf(v0 - m);
        if (lane + 32 < C) s += expf(v1 - m);
        #pragma unroll
        for (int off = 16; off; off >>= 1)
            s += __shfl_xor(s, off, 32);
        float ls = logf(s);
        if (lane      < C) p[lane]      = v0 - m - ls;
        if (lane + 32 < C) p[lane + 32] = v1 - m - ls;
    }
}

// ---------------------------------------------------------------------------
extern "C" void kernel_launch(void* const* d_in, const int* in_sizes, int n_in,
                              void* d_out, int out_size, void* d_ws, size_t ws_size,
                              hipStream_t stream)
{
    const float* x  = (const float*)d_in[0];
    const int*   ei = (const int*)  d_in[1];
    const float* ew = (const float*)d_in[2];
    const float* W1 = (const float*)d_in[3];
    const float* b1 = (const float*)d_in[4];
    const float* W2 = (const float*)d_in[5];
    const float* b2 = (const float*)d_in[6];
    const float* W3 = (const float*)d_in[7];
    const float* b3 = (const float*)d_in[8];
    const float* Wl = (const float*)d_in[9];
    const float* bl = (const float*)d_in[10];

    const int H    = in_sizes[4];              // 256
    const int F_IN = in_sizes[3] / H;          // 512
    const int Nn   = in_sizes[0] / F_IN;       // 50000 nodes
    const int E    = in_sizes[2];              // 800000 edges
    const int OUT  = in_sizes[8];              // 256
    const int Cc   = in_sizes[10];             // 40 classes
    const int KT   = 2 * H + OUT;              // 768 concat width

    const int* src = ei;
    const int* dst = ei + E;

    // Workspace carve-up
    float* Z     = (float*)d_ws;               // Nn x KT  (x1|x2|x3 columns)
    float* Htmp  = Z    + (long)Nn * KT;       // Nn x max(H,OUT)
    float* Agg   = Htmp + (long)Nn * H;        // Nn x max(H,OUT)
    float* degw  = Agg  + (long)Nn * H;
    float* deg1  = degw + Nn;
    float* dinvw = deg1 + Nn;
    float* dinv1 = dinvw + Nn;
    float* logits = (float*)d_out;             // Nn x Cc

    dim3 blk128(128), blk256(256);
    const int zgrid  = 4096;
    const int aggrid = 2048;                   // 16384 waves, grid-stride over E

    // Degrees + normalization (re-zero every call: ws is not re-poisoned)
    gcn_fill_zero<<<64, blk256, 0, stream>>>(degw, (long)2 * Nn);
    gcn_degree<<<1024, blk256, 0, stream>>>(dst, ew, E, degw, deg1);
    gcn_dinv<<<(Nn + 255) / 256, blk256, 0, stream>>>(degw, deg1, dinvw, dinv1, Nn);

    dim3 g1((H + TILE_N - 1) / TILE_N, (Nn + TILE_M - 1) / TILE_M);

    // ---- Layer 1: x1 = relu(norm-agg(x @ W1) + b1) -> Z[:, 0:H]
    gcn_fill_zero<<<zgrid, blk256, 0, stream>>>(Agg, (long)Nn * H);
    gcn_gemm_wmma<<<g1, blk128, 0, stream>>>(x, F_IN, W1, nullptr, Htmp, H, Nn, H, F_IN);
    gcn_aggregate<<<aggrid, blk256, 0, stream>>>(Htmp, src, dst, ew, dinvw, Agg, H, E);
    gcn_finalize<<<zgrid, blk256, 0, stream>>>(Agg, Htmp, dinvw, b1, Z, KT, H, Nn, 1);

    // ---- Layer 2: x2 = relu(norm-agg(x1 @ W2) + b2) -> Z[:, H:2H]
    gcn_fill_zero<<<zgrid, blk256, 0, stream>>>(Agg, (long)Nn * H);
    gcn_gemm_wmma<<<g1, blk128, 0, stream>>>(Z, KT, W2, nullptr, Htmp, H, Nn, H, H);
    gcn_aggregate<<<aggrid, blk256, 0, stream>>>(Htmp, src, dst, ew, dinvw, Agg, H, E);
    gcn_finalize<<<zgrid, blk256, 0, stream>>>(Agg, Htmp, dinvw, b2, Z + H, KT, H, Nn, 1);

    // ---- Layer 3 (unit edge weights): x3 = norm1-agg(x2 @ W3) + b3 -> Z[:, 2H:]
    dim3 g3((OUT + TILE_N - 1) / TILE_N, (Nn + TILE_M - 1) / TILE_M);
    gcn_fill_zero<<<zgrid, blk256, 0, stream>>>(Agg, (long)Nn * OUT);
    gcn_gemm_wmma<<<g3, blk128, 0, stream>>>(Z + H, KT, W3, nullptr, Htmp, OUT, Nn, OUT, H);
    gcn_aggregate<<<aggrid, blk256, 0, stream>>>(Htmp, src, dst, nullptr, dinv1, Agg, OUT, E);
    gcn_finalize<<<zgrid, blk256, 0, stream>>>(Agg, Htmp, dinv1, b3, Z + 2 * H, KT, OUT, Nn, 0);

    // ---- Head: logits = Z @ Wl + bl, then log_softmax rows
    dim3 gl((Cc + TILE_N - 1) / TILE_N, (Nn + TILE_M - 1) / TILE_M);
    gcn_gemm_wmma<<<gl, blk128, 0, stream>>>(Z, KT, Wl, bl, logits, Cc, Nn, Cc, KT);
    gcn_log_softmax<<<(Nn * 32 + 255) / 256, blk256, 0, stream>>>(logits, Nn, Cc);
}
